// LLELayer_64055142252570
// MI455X (gfx1250) — matrix operand order repaired and checked by reading, not compiled, not run on hardware
//
#include <hip/hip_runtime.h>
#include <math.h>

typedef __attribute__((ext_vector_type(16))) _Float16 v16h;
typedef __attribute__((ext_vector_type(8)))  _Float16 v8h;
typedef __attribute__((ext_vector_type(8)))  float    v8f;

#define NPTS 4096
#define NF   256
#define NK   16
#define NDIM 8
#define REGC 0.001f
#define ITERS 200

// ---------------------------------------------------------------------------
// 1) Convert X -> f16 and compute row squared norms
// ---------------------------------------------------------------------------
__global__ void k_convert_sq(const float* __restrict__ X, _Float16* __restrict__ Xh,
                             float* __restrict__ sq) {
  int i = blockIdx.x, t = threadIdx.x;
  float v = X[(size_t)i * NF + t];
  Xh[(size_t)i * NF + t] = (_Float16)v;
  __shared__ float red[256];
  red[t] = v * v;
  __syncthreads();
  for (int s = 128; s > 0; s >>= 1) { if (t < s) red[t] += red[t + s]; __syncthreads(); }
  if (t == 0) sq[i] = red[0];
}

// ---------------------------------------------------------------------------
// 2) D2 = sq_i + sq_j - 2 X X^T via WMMA f32_16x16x32_f16.  One wave = one
//    16x16 tile; 8 waves per 256-thread block.
// ---------------------------------------------------------------------------
__global__ void k_dist(const _Float16* __restrict__ Xh, const float* __restrict__ sq,
                       float* __restrict__ D2) {
  const int lane = threadIdx.x & 31;
  const int wave = threadIdx.x >> 5;
  const int tm = blockIdx.x * 8 + wave;   // 0..255
  const int tn = blockIdx.y;              // 0..255
  const int half = lane >> 4;             // 0 or 1
  const int l15  = lane & 15;
  const int rowA = tm * 16 + l15;         // A-matrix row (M)
  const int colB = tn * 16 + l15;         // B-matrix col (N)
  const int akb  = half ? 16 : 0;         // A K-base per half-wave
  const int bkb  = half ? 16 : 0;         // B K-base per half-wave

  v8f acc = {0.f, 0.f, 0.f, 0.f, 0.f, 0.f, 0.f, 0.f};

  for (int k0 = 0; k0 < NF; k0 += 32) {
    // A: lane holds K = akb + {0..7} and akb + 8 + {0..7}  (two 16B chunks)
    v8h a0 = *(const v8h*)(Xh + (size_t)rowA * NF + k0 + akb);
    v8h a1 = *(const v8h*)(Xh + (size_t)rowA * NF + k0 + akb + 8);
    // B[k][n] = X[colB][k]; lane holds K = bkb + {0..15} contiguous (32B)
    v16h b = *(const v16h*)(Xh + (size_t)colB * NF + k0 + bkb);
    __builtin_prefetch(Xh + (size_t)rowA * NF + k0 + 32, 0, 1);

    v16h a;
#pragma unroll
    for (int e = 0; e < 8; ++e) { a[e] = a0[e]; a[e + 8] = a1[e]; }

    acc = __builtin_amdgcn_wmma_f32_16x16x32_f16(
        /*neg_a=*/false, a, /*neg_b=*/false, b,
        /*c_mod=*/(short)0, acc, /*reuse_a=*/false, /*reuse_b=*/false);
  }

  const int col = tn * 16 + l15;
#pragma unroll
  for (int r = 0; r < 8; ++r) {
    int row = tm * 16 + r + (half ? 8 : 0);
    float d = sq[row] + sq[col] - 2.0f * acc[r];
    if (row == col) d = __builtin_inff();
    D2[(size_t)row * NPTS + col] = d;
  }
}

// ---------------------------------------------------------------------------
// 3) Top-16 smallest distances per row (iterated argmin in LDS)
// ---------------------------------------------------------------------------
__global__ void k_topk(const float* __restrict__ D2, int* __restrict__ nidx) {
  int i = blockIdx.x, t = threadIdx.x;
  __shared__ float d[NPTS];      // 16 KB
  __shared__ float rv[256];
  __shared__ int   ri[256];
  for (int j = t; j < NPTS; j += 256) d[j] = D2[(size_t)i * NPTS + j];
  __syncthreads();
  for (int s = 0; s < NK; ++s) {
    float bv = __builtin_inff(); int bi = 0;
    for (int j = t; j < NPTS; j += 256) { float v = d[j]; if (v < bv) { bv = v; bi = j; } }
    rv[t] = bv; ri[t] = bi;
    __syncthreads();
    for (int h = 128; h > 0; h >>= 1) {
      if (t < h && rv[t + h] < rv[t]) { rv[t] = rv[t + h]; ri[t] = ri[t + h]; }
      __syncthreads();
    }
    if (t == 0) { nidx[i * NK + s] = ri[0]; d[ri[0]] = __builtin_inff(); }
    __syncthreads();
  }
}

// ---------------------------------------------------------------------------
// 4) Per-point: center neighbors, Gram C (+reg I), solve C w = 1, normalize
// ---------------------------------------------------------------------------
__global__ void k_weights(const float* __restrict__ X, const int* __restrict__ nidx,
                          float* __restrict__ wgt) {
  int i = blockIdx.x, t = threadIdx.x;
  __shared__ float cen[NK * 257];     // padded stride to avoid bank conflicts
  __shared__ float Cm[NK][NK + 1];
  __shared__ float wsh[NK];

  int j  = t >> 4;          // neighbor 0..15
  int f0 = (t & 15) * 16;   // feature chunk
  int nb = nidx[i * NK + j];
  for (int f = f0; f < f0 + 16; ++f)
    cen[j * 257 + f] = X[(size_t)nb * NF + f] - X[(size_t)i * NF + f];
  __syncthreads();

  int l = t & 15;
  float acc = 0.f;
  for (int f = 0; f < NF; ++f) acc += cen[j * 257 + f] * cen[l * 257 + f];
  if (j == l) acc += REGC;
  Cm[j][l] = acc;
  __syncthreads();

  if (t == 0) {
    float rhs[NK];
    for (int r = 0; r < NK; ++r) rhs[r] = 1.f;
    for (int p = 0; p < NK; ++p) {
      float piv = Cm[p][p];
      if (fabsf(piv) < 1e-12f) piv = 1e-12f;
      float inv = 1.f / piv;
      for (int c = p; c < NK; ++c) Cm[p][c] *= inv;
      rhs[p] *= inv;
      for (int r = p + 1; r < NK; ++r) {
        float fct = Cm[r][p];
        for (int c = p; c < NK; ++c) Cm[r][c] -= fct * Cm[p][c];
        rhs[r] -= fct * rhs[p];
      }
    }
    float sum = 0.f;
    for (int r = NK - 1; r >= 0; --r) {
      float s = rhs[r];
      for (int c = r + 1; c < NK; ++c) s -= Cm[r][c] * wsh[c];
      wsh[r] = s;
    }
    for (int r = 0; r < NK; ++r) sum += wsh[r];
    if (fabsf(sum) < 1e-20f) sum = 1e-20f;
    float inv = 1.f / sum;
    for (int r = 0; r < NK; ++r) wsh[r] *= inv;
  }
  __syncthreads();
  if (t < NK) wgt[i * NK + t] = wsh[t];
}

// ---------------------------------------------------------------------------
// 5) M = I - W - W^T + W^T W  (M aliases the D2 buffer)
// ---------------------------------------------------------------------------
__global__ void k_initM(float* __restrict__ M) {
  size_t total = (size_t)NPTS * NPTS;
  for (size_t p = (size_t)blockIdx.x * blockDim.x + threadIdx.x; p < total;
       p += (size_t)gridDim.x * blockDim.x)
    M[p] = ((p / NPTS) == (p % NPTS)) ? 1.f : 0.f;
}

__global__ void k_zero_scalars(float* __restrict__ scal) {
  if (threadIdx.x < 16) scal[threadIdx.x] = 0.f;
}

__global__ void k_scatterM(const int* __restrict__ nidx, const float* __restrict__ wgt,
                           float* __restrict__ M) {
  int n = blockIdx.x, t = threadIdx.x;
  __shared__ int   sid[NK];
  __shared__ float sw[NK];
  if (t < NK) { sid[t] = nidx[n * NK + t]; sw[t] = wgt[n * NK + t]; }
  __syncthreads();
  int j = t >> 4, l = t & 15;
  atomicAdd(&M[(size_t)sid[j] * NPTS + sid[l]], sw[j] * sw[l]);
  if (t < NK) {
    atomicAdd(&M[(size_t)n * NPTS + sid[t]], -sw[t]);
    atomicAdd(&M[(size_t)sid[t] * NPTS + n], -sw[t]);
  }
}

// ---------------------------------------------------------------------------
// 6) Gershgorin bound: shift = max_i sum_j |M_ij|  (>= lambda_max, all >= 0)
// ---------------------------------------------------------------------------
__global__ void k_gershgorin(const float* __restrict__ M, float* __restrict__ scal) {
  int i = blockIdx.x, t = threadIdx.x;
  float s = 0.f;
  const float* row = M + (size_t)i * NPTS;
  for (int j = t; j < NPTS; j += 256) s += fabsf(row[j]);
  __shared__ float red[256];
  red[t] = s;
  __syncthreads();
  for (int h = 128; h > 0; h >>= 1) { if (t < h) red[t] += red[t + h]; __syncthreads(); }
  if (t == 0) atomicMax((int*)scal, __float_as_int(red[0]));  // non-negative floats
}

// ---------------------------------------------------------------------------
// 7) Subspace iteration on B = shift*I - M.  Q[0] = constant null-vector.
// ---------------------------------------------------------------------------
__global__ void k_initQ(float* __restrict__ Q) {
  int g = blockIdx.x * blockDim.x + threadIdx.x;
  if (g < 9 * NPTS) {
    int v = g >> 12;
    float val;
    if (v == 0) {
      val = 0.015625f;  // 1/sqrt(4096)
    } else {
      unsigned h = (unsigned)g * 2654435761u ^ (unsigned)v * 0x9E3779B9u;
      h ^= h >> 16; h *= 0x7feb352du; h ^= h >> 15;
      val = ((float)(h & 0xFFFFFFu) / 8388608.0f) - 1.0f;
    }
    Q[g] = val;
  }
}

__global__ void k_matvec(const float* __restrict__ M, const float* __restrict__ Q,
                         float* __restrict__ Y, const float* __restrict__ scal) {
  int i = blockIdx.x, t = threadIdx.x;
  float acc[8] = {0.f, 0.f, 0.f, 0.f, 0.f, 0.f, 0.f, 0.f};
  const float* row = M + (size_t)i * NPTS;
  for (int j = t; j < NPTS; j += 256) {
    float m = row[j];
#pragma unroll
    for (int v = 0; v < 8; ++v) acc[v] += m * Q[(size_t)(v + 1) * NPTS + j];
  }
  __shared__ float red[8][256];
#pragma unroll
  for (int v = 0; v < 8; ++v) red[v][t] = acc[v];
  __syncthreads();
  for (int h = 128; h > 0; h >>= 1) {
    if (t < h)
#pragma unroll
      for (int v = 0; v < 8; ++v) red[v][t] += red[v][t + h];
    __syncthreads();
  }
  if (t == 0) {
    float shift = scal[0];
#pragma unroll
    for (int v = 0; v < 8; ++v)
      Y[(size_t)(v + 1) * NPTS + i] = shift * Q[(size_t)(v + 1) * NPTS + i] - red[v][0];
  }
}

// Single-block Gram-Schmidt: orthonormalize Y[1..8] against Q[0..v-1] -> Q[1..8]
__global__ void k_gs(float* __restrict__ Q, float* __restrict__ Y) {
  int t = threadIdx.x;
  __shared__ float red[256];
  __shared__ float sval;
  for (int v = 1; v <= 8; ++v) {
    for (int u = 0; u < v; ++u) {
      float p = 0.f;
      for (int i = t; i < NPTS; i += 256) p += Y[(size_t)v * NPTS + i] * Q[(size_t)u * NPTS + i];
      red[t] = p; __syncthreads();
      for (int h = 128; h > 0; h >>= 1) { if (t < h) red[t] += red[t + h]; __syncthreads(); }
      if (t == 0) sval = red[0];
      __syncthreads();
      float dd = sval;
      for (int i = t; i < NPTS; i += 256) Y[(size_t)v * NPTS + i] -= dd * Q[(size_t)u * NPTS + i];
      __syncthreads();
    }
    float p = 0.f;
    for (int i = t; i < NPTS; i += 256) { float y = Y[(size_t)v * NPTS + i]; p += y * y; }
    red[t] = p; __syncthreads();
    for (int h = 128; h > 0; h >>= 1) { if (t < h) red[t] += red[t + h]; __syncthreads(); }
    if (t == 0) sval = red[0];
    __syncthreads();
    float inv = rsqrtf(fmaxf(sval, 1e-30f));
    for (int i = t; i < NPTS; i += 256) Q[(size_t)v * NPTS + i] = Y[(size_t)v * NPTS + i] * inv;
    __syncthreads();
  }
}

// ---------------------------------------------------------------------------
// 8) Rayleigh quotients lam[v] = q_v^T M q_v, then sort + emit embedding
// ---------------------------------------------------------------------------
__global__ void k_rayleigh(const float* __restrict__ M, const float* __restrict__ Q,
                           float* __restrict__ scal) {
  int i = blockIdx.x, t = threadIdx.x;
  float acc[8] = {0.f, 0.f, 0.f, 0.f, 0.f, 0.f, 0.f, 0.f};
  const float* row = M + (size_t)i * NPTS;
  for (int j = t; j < NPTS; j += 256) {
    float m = row[j];
#pragma unroll
    for (int v = 0; v < 8; ++v) acc[v] += m * Q[(size_t)(v + 1) * NPTS + j];
  }
  __shared__ float red[8][256];
#pragma unroll
  for (int v = 0; v < 8; ++v) red[v][t] = acc[v];
  __syncthreads();
  for (int h = 128; h > 0; h >>= 1) {
    if (t < h)
#pragma unroll
      for (int v = 0; v < 8; ++v) red[v][t] += red[v][t + h];
    __syncthreads();
  }
  if (t == 0) {
#pragma unroll
    for (int v = 0; v < 8; ++v)
      atomicAdd(&scal[1 + v], Q[(size_t)(v + 1) * NPTS + i] * red[v][0]);
  }
}

__global__ void k_final(const float* __restrict__ Q, const float* __restrict__ scal,
                        float* __restrict__ out) {
  __shared__ int perm[8];
  if (threadIdx.x == 0) {
    int p[8]; float lv[8];
    for (int c = 0; c < 8; ++c) { p[c] = c + 1; lv[c] = scal[c + 1]; }
    for (int a = 1; a < 8; ++a) {           // insertion sort ascending
      int pa = p[a]; float la = lv[a]; int b = a - 1;
      while (b >= 0 && lv[b] > la) { lv[b + 1] = lv[b]; p[b + 1] = p[b]; --b; }
      lv[b + 1] = la; p[b + 1] = pa;
    }
    for (int c = 0; c < 8; ++c) perm[c] = p[c];
  }
  __syncthreads();
  for (int i = threadIdx.x; i < NPTS; i += 256)
#pragma unroll
    for (int c = 0; c < 8; ++c)
      out[(size_t)i * NDIM + c] = Q[(size_t)perm[c] * NPTS + i];
}

// ---------------------------------------------------------------------------
extern "C" void kernel_launch(void* const* d_in, const int* in_sizes, int n_in,
                              void* d_out, int out_size, void* d_ws, size_t ws_size,
                              hipStream_t stream) {
  const float* X = (const float*)d_in[0];
  float* out = (float*)d_out;

  char* base = (char*)d_ws;
  size_t off = 0;
  auto alloc = [&](size_t b) -> char* {
    char* r = base + off;
    off += (b + 255) & ~(size_t)255;
    return r;
  };
  float*    sq   = (float*)alloc((size_t)NPTS * 4);
  _Float16* Xh   = (_Float16*)alloc((size_t)NPTS * NF * 2);
  int*      nidx = (int*)alloc((size_t)NPTS * NK * 4);
  float*    wgt  = (float*)alloc((size_t)NPTS * NK * 4);
  float*    scal = (float*)alloc(64);                       // [0]=shift, [1..8]=lam
  float*    Q    = (float*)alloc((size_t)9 * NPTS * 4);
  float*    Y    = (float*)alloc((size_t)9 * NPTS * 4);
  float*    D2M  = (float*)alloc((size_t)NPTS * NPTS * 4);  // D2, later reused as M

  k_convert_sq<<<NPTS, 256, 0, stream>>>(X, Xh, sq);
  k_dist<<<dim3(32, 256), 256, 0, stream>>>(Xh, sq, D2M);
  k_topk<<<NPTS, 256, 0, stream>>>(D2M, nidx);
  k_weights<<<NPTS, 256, 0, stream>>>(X, nidx, wgt);

  k_initM<<<8192, 256, 0, stream>>>(D2M);                   // D2 dead -> becomes M
  k_zero_scalars<<<1, 32, 0, stream>>>(scal);
  k_scatterM<<<NPTS, 256, 0, stream>>>(nidx, wgt, D2M);
  k_gershgorin<<<NPTS, 256, 0, stream>>>(D2M, scal);
  k_initQ<<<(9 * NPTS + 255) / 256, 256, 0, stream>>>(Q);

  for (int it = 0; it < ITERS; ++it) {
    k_matvec<<<NPTS, 256, 0, stream>>>(D2M, Q, Y, scal);
    k_gs<<<1, 256, 0, stream>>>(Q, Y);
  }

  k_rayleigh<<<NPTS, 256, 0, stream>>>(D2M, Q, scal);
  k_final<<<1, 256, 0, stream>>>(Q, scal, out);
}
